// BaseObservedLayer_45157286150244
// MI455X (gfx1250) — matrix operand order repaired and checked by reading, not compiled, not run on hardware
//
#include <hip/hip_runtime.h>

#define B_DIM 2048
#define F_DIM 4096
#define K_DIM 64

typedef __attribute__((ext_vector_type(16))) _Float16 v16h;
typedef __attribute__((ext_vector_type(8)))  float    v8f;

// ---------------- stateless counter RNG ----------------
__device__ __forceinline__ unsigned hashu(unsigned x) {
    x ^= x >> 16; x *= 0x7feb352dU;
    x ^= x >> 15; x *= 0x846ca68bU;
    x ^= x >> 16;
    return x;
}
__device__ __forceinline__ float u01(unsigned h) {
    return (float)(h >> 8) * (1.0f / 16777216.0f) + (0.5f / 16777216.0f);
}
// popcount normal: popc(h) ~ Binomial(32, 1/2), mean 16, var 8.
// z = (popc - 16)/sqrt(8), plus 8-bit dither to smooth the lattice.
// One hash + v_popcnt + fma: ~3 VALU ops vs ~28 for a 3-uniform normal.
__device__ __forceinline__ float zpop(unsigned h) {
    float zi = (float)__popc(h) - 16.0f;
    float dither = (float)(h & 0xffu) * (1.0f / 256.0f) - 0.5f;
    return (zi + dither) * 0.35355339f;   // 1/sqrt(8)
}
// Box-Muller normal with fast-math trans ops (Gamma sampling only)
__device__ __forceinline__ float zbm(unsigned s) {
    float u1 = u01(hashu(s ^ 0x9e3779b1u));
    float u2 = u01(hashu(s ^ 0x85ebca6bu));
    return __builtin_amdgcn_sqrtf(-2.0f * __logf(u1)) * __cosf(6.28318530718f * u2);
}

// ---------------- kernel 0: zero an fp32 buffer ----------------
__global__ void zero_f32(float* p, int n) {
    int i = blockIdx.x * blockDim.x + threadIdx.x;
    if (i < n) p[i] = 0.0f;
}

// ---------------- kernel 1: row-normalize phi (K rows of F) ----------------
__global__ __launch_bounds__(256) void normalize_phi(const float* __restrict__ phi,
                                                     float* __restrict__ phi_n) {
    const int k = blockIdx.x;
    const int tid = threadIdx.x;
    __shared__ float red[256];
    float s = 0.0f;
    for (int j = tid; j < F_DIM; j += 256) s += phi[k * F_DIM + j];
    red[tid] = s;
    __syncthreads();
    for (int off = 128; off > 0; off >>= 1) {
        if (tid < off) red[tid] += red[tid + off];
        __syncthreads();
    }
    const float inv = 1.0f / red[0];
    for (int j = tid; j < F_DIM; j += 256)
        phi_n[k * F_DIM + j] = phi[k * F_DIM + j] * inv;
}

// ---------------- kernel 2: fused WMMA denom + multinomial thinning ----------------
// block tile: 16 rows (i) x 128 cols (j); 8 waves, each wave owns a 16x16 sub-tile.
__global__ __launch_bounds__(256) void sampler_wmma(const int*   __restrict__ x,
                                                    const float* __restrict__ theta,
                                                    const float* __restrict__ phi_n,
                                                    float* __restrict__ m_out,   // (B,K) atomic
                                                    float* __restrict__ xkf) {   // (K,F) atomic
    const int jbase = blockIdx.x * 128;
    const int i0    = blockIdx.y * 16;
    const int tid   = threadIdx.x;
    const int wave  = tid >> 5;       // 0..7 (wave32)
    const int lane  = tid & 31;
    const int n     = lane & 15;      // column within 16-wide sub-tile
    const int half  = lane >> 4;      // 0/1

    __shared__ _Float16 thh[16][64];    // theta tile  (i,k)  f16 for WMMA A
    __shared__ _Float16 phh[64][128];   // phi tile    (k,j)  f16 for WMMA B
    __shared__ float    m_tile[16 * 64];
    __shared__ float    x_tile[128 * 64];

    for (int t = tid; t < 16 * 64; t += 256) m_tile[t] = 0.0f;

    // stage theta tile (16x64)
    for (int t = tid; t < 16 * 64; t += 256) {
        int ii = t >> 6, kk = t & 63;
        thh[ii][kk] = (_Float16)theta[(i0 + ii) * K_DIM + kk];
    }
    // stage phi tile (64x128), coalesced along j
    for (int t = tid; t < 64 * 128; t += 256) {
        int kk = t >> 7, jj = t & 127;
        phh[kk][jj] = (_Float16)phi_n[kk * F_DIM + jbase + jj];
    }
    __syncthreads();

    const int jloc  = wave * 16 + n;       // column within block tile
    const int jglob = jbase + jloc;

    // prefetch the count tile we are about to consume
    __builtin_prefetch(&x[(i0 + half * 8) * F_DIM + jglob], 0, 0);

    // ---- denom tile via two chained v_wmma_f32_16x16x32_f16 (K = 64) ----
    v8f c = {};
    #pragma unroll
    for (int chain = 0; chain < 2; ++chain) {
        v16h a, b;
        #pragma unroll
        for (int v = 0; v < 8; ++v) {
            // A fragment (16x32 f16): lane holds row M = n
            int ka = 32 * chain + ((v < 4) ? 0 : 16) + 8 * half + 2 * (v & 3);
            a[2 * v]     = thh[n][ka];
            a[2 * v + 1] = thh[n][ka + 1];
            // B fragment (32x16 f16): lane holds column N = n
            int kb = 32 * chain + ((v < 4) ? 0 : 16) + 8 * half + 2 * (v & 3);
            b[2 * v]     = phh[kb][jloc];
            b[2 * v + 1] = phh[kb + 1][jloc];
        }
        c = __builtin_amdgcn_wmma_f32_16x16x32_f16(
                /*neg_a=*/false, a, /*neg_b=*/false, b,
                /*c_mod=*/(short)0, c, /*reuse_a=*/false, /*reuse_b=*/false);
    }

    // ---- per-thread thinning chain on the 8 elements this thread owns ----
    float nrem[8], denv[8];
    int   iloc[8];
    #pragma unroll
    for (int v = 0; v < 8; ++v) {
        iloc[v] = v + 8 * half;                                  // C/D layout row
        nrem[v] = (float)x[(i0 + iloc[v]) * F_DIM + jglob];
        denv[v] = c[v];
    }

    const unsigned base_seed = (unsigned)(i0 * 131071u) ^ (unsigned)(jglob * 2654435761u);

    for (int k = 0; k < K_DIM; ++k) {
        const float ph = (float)phh[k][jloc];
        const unsigned ks = base_seed ^ (unsigned)(k * 0xc2b2ae35u);
        float xsum = 0.0f;
        #pragma unroll
        for (int v = 0; v < 8; ++v) {
            const float th = (float)thh[iloc[v]][k];
            const float p  = th * ph;
            // fast reciprocal (v_rcp_f32) instead of IEEE division
            float ratio = p * __builtin_amdgcn_rcpf(fmaxf(denv[v], 1e-30f));
            ratio = fminf(fmaxf(ratio, 0.0f), 1.0f);
            const float nn   = nrem[v];
            const float mean = nn * ratio;
            const float var  = fmaxf(mean * (1.0f - ratio), 0.0f);
            // single hash per sample; normal from popcount
            const unsigned s = hashu(ks + (unsigned)(iloc[v] * 0x27d4eb2fu));
            float chi = __builtin_roundf(mean + __builtin_amdgcn_sqrtf(var) * zpop(s));
            chi = fminf(fmaxf(chi, 0.0f), nn);
            nrem[v] = nn - chi;
            denv[v] -= p;
            xsum += chi;

            // butterfly-reduce chi over the 16 lanes sharing (iloc[v], k):
            // masks 1,2,4,8 stay within each 16-lane half of the wave32
            float msum = chi;
            msum += __shfl_xor(msum, 1, 32);
            msum += __shfl_xor(msum, 2, 32);
            msum += __shfl_xor(msum, 4, 32);
            msum += __shfl_xor(msum, 8, 32);
            if (n == 0) atomicAdd(&m_tile[iloc[v] * 64 + k], msum);  // 2 lanes/wave
        }
        // combine the two halves that share this jloc; then exactly one
        // writer per (jloc, k) in the whole block -> plain store, no atomic
        xsum += __shfl_xor(xsum, 16, 32);
        if (half == 0) x_tile[jloc * 64 + k] = xsum;
    }
    __syncthreads();

    // flush block tiles with global atomics
    for (int t = tid; t < 16 * 64; t += 256) {
        int ii = t >> 6, kk = t & 63;
        float v = m_tile[t];
        if (v != 0.0f) atomicAdd(&m_out[(i0 + ii) * K_DIM + kk], v);
    }
    for (int t = tid; t < 128 * 64; t += 256) {
        int jj = t >> 6, kk = t & 63;
        float v = x_tile[t];
        if (v != 0.0f) atomicAdd(&xkf[kk * F_DIM + jbase + jj], v);
    }
}

// ---------------- kernel 3: Dirichlet rows phi_new[k,:] ~ Dir(xkf[k,:] + 1) ----------------
__device__ __forceinline__ float gamma_sample(float alpha, unsigned seed) {
    // Marsaglia-Tsang, alpha >= 1 (eta = 1 so alpha = count + 1)
    const float d = alpha - (1.0f / 3.0f);
    const float cc = __builtin_amdgcn_rsqf(9.0f * d);
    #pragma unroll 1
    for (int it = 0; it < 8; ++it) {
        unsigned s = hashu(seed + (unsigned)it * 0x9e3779b9u);
        float z = zbm(s);
        float v = 1.0f + cc * z;
        if (v <= 0.0f) continue;
        v = v * v * v;
        float u = u01(hashu(s ^ 0xdeadbeefu));
        float z2 = z * z;
        if (u < 1.0f - 0.0331f * z2 * z2) return d * v;
        if (__logf(u) < 0.5f * z2 + d * (1.0f - v + __logf(v))) return d * v;
    }
    return d; // deterministic fallback near the mode
}

__global__ __launch_bounds__(256) void dirichlet_rows(const float* __restrict__ xkf,
                                                      float* __restrict__ phi_out) {
    const int k   = blockIdx.x;
    const int tid = threadIdx.x;
    __shared__ float red[256];
    float g[F_DIM / 256];
    float s = 0.0f;
    #pragma unroll
    for (int t = 0; t < F_DIM / 256; ++t) {
        const int j = tid + t * 256;
        const float alpha = xkf[k * F_DIM + j] + 1.0f;  // ETA = 1
        const unsigned seed = hashu((unsigned)(k * 0x85ebca6bu) ^ (unsigned)(j * 0xc2b2ae35u));
        const float gg = gamma_sample(alpha, seed);
        g[t] = gg;
        s += gg;
    }
    red[tid] = s;
    __syncthreads();
    for (int off = 128; off > 0; off >>= 1) {
        if (tid < off) red[tid] += red[tid + off];
        __syncthreads();
    }
    const float inv = 1.0f / red[0];
    #pragma unroll
    for (int t = 0; t < F_DIM / 256; ++t) {
        const int j = tid + t * 256;
        phi_out[k * F_DIM + j] = g[t] * inv;
    }
}

// ---------------- host launcher ----------------
extern "C" void kernel_launch(void* const* d_in, const int* in_sizes, int n_in,
                              void* d_out, int out_size, void* d_ws, size_t ws_size,
                              hipStream_t stream) {
    const int*   x     = (const int*)d_in[0];    // (B,F) int32
    const float* theta = (const float*)d_in[1];  // (B,K)
    const float* phi   = (const float*)d_in[2];  // (K,F)

    float* out_m   = (float*)d_out;                       // (B,K)  = 131072 floats
    float* out_phi = (float*)d_out + B_DIM * K_DIM;       // (K,F)  = 262144 floats

    // workspace: phi_n (K*F) | xkf (K*F)
    float* phi_n = (float*)d_ws;
    float* xkf   = phi_n + K_DIM * F_DIM;

    // zero atomic accumulators (d_out is poisoned by the harness)
    {
        int n1 = B_DIM * K_DIM;
        zero_f32<<<(n1 + 255) / 256, 256, 0, stream>>>(out_m, n1);
        int n2 = K_DIM * F_DIM;
        zero_f32<<<(n2 + 255) / 256, 256, 0, stream>>>(xkf, n2);
    }

    // 1) phi row normalization
    normalize_phi<<<K_DIM, 256, 0, stream>>>(phi, phi_n);

    // 2) fused WMMA denominator + thinning sampler
    dim3 grid(F_DIM / 128, B_DIM / 16);
    sampler_wmma<<<grid, 256, 0, stream>>>(x, theta, phi_n, out_m, xkf);

    // 3) Dirichlet resampling of phi rows
    dirichlet_rows<<<K_DIM, 256, 0, stream>>>(xkf, out_phi);
}
